// MoE_58884001628642
// MI455X (gfx1250) — compile-verified, matrix operand order checked by reading, not run on hardware
//
#include <hip/hip_runtime.h>

#define N_TOKENS 32768
#define DIM      1024
#define NEXP     8

#define BM 128
#define BN 128
#define BK 32
#define LDK 40   // padded K stride in bf16 elements (80 bytes -> 16B aligned rows)

typedef __attribute__((ext_vector_type(16))) __bf16 v16bf;
typedef __attribute__((ext_vector_type(8)))  __bf16 v8bf;
typedef __attribute__((ext_vector_type(4)))  __bf16 v4bf;
typedef __attribute__((ext_vector_type(8)))  float  v8f;

// ---------------------------------------------------------------- zero output
__global__ void moe_zero_kernel(float4* __restrict__ out, int n4) {
    int i = blockIdx.x * blockDim.x + threadIdx.x;
    int stride = gridDim.x * blockDim.x;
    float4 z = {0.f, 0.f, 0.f, 0.f};
    for (; i < n4; i += stride) out[i] = z;
}

// ---------------------------------------------------------------- gating
// One wave32 per token: gate scores, softmax, top-2, weight = sum of top-2.
// Appends token to the two selected experts' compacted lists.
__global__ __launch_bounds__(256) void moe_gate_kernel(
    const float* __restrict__ x, const float* __restrict__ Wg,
    const float* __restrict__ bg, float* __restrict__ wgt,
    int* __restrict__ counts, int* __restrict__ lists) {
    const int lane  = threadIdx.x & 31;
    const int wave  = threadIdx.x >> 5;
    const int token = blockIdx.x * 8 + wave;
    const float* xr = x + (size_t)token * DIM;

    float acc[NEXP];
#pragma unroll
    for (int e = 0; e < NEXP; ++e) acc[e] = 0.f;
    for (int d = lane; d < DIM; d += 32) {
        float xv = xr[d];
        const float* wr = Wg + d * NEXP;
#pragma unroll
        for (int e = 0; e < NEXP; ++e) acc[e] += xv * wr[e];
    }
#pragma unroll
    for (int e = 0; e < NEXP; ++e) {
#pragma unroll
        for (int off = 16; off > 0; off >>= 1)
            acc[e] += __shfl_xor(acc[e], off, 32);
    }
    if (lane == 0) {
        float s[NEXP];
        float m = -1e30f;
#pragma unroll
        for (int e = 0; e < NEXP; ++e) { s[e] = acc[e] + bg[e]; m = fmaxf(m, s[e]); }
        float sum = 0.f;
#pragma unroll
        for (int e = 0; e < NEXP; ++e) { s[e] = expf(s[e] - m); sum += s[e]; }
        float inv = 1.f / sum;
#pragma unroll
        for (int e = 0; e < NEXP; ++e) s[e] *= inv;
        int i1 = 0;
#pragma unroll
        for (int e = 1; e < NEXP; ++e) if (s[e] > s[i1]) i1 = e;   // strict > : first-index tie rule
        int i2 = (i1 == 0) ? 1 : 0;
#pragma unroll
        for (int e = 0; e < NEXP; ++e) if (e != i1 && s[e] > s[i2]) i2 = e;

        wgt[token] = s[i1] + s[i2];
        int p1 = atomicAdd(&counts[i1], 1);
        lists[i1 * N_TOKENS + p1] = token;
        int p2 = atomicAdd(&counts[i2], 1);
        lists[i2 * N_TOKENS + p2] = token;
    }
}

// ---------------------------------------------------------------- expert GEMM
// Per expert, compacted rows. 128x128x32 block tile, 8 waves (2x4), each wave
// computes a 64x32 sub-tile = 4x2 WMMA fragments. Split-bf16 (hi+lo) inputs:
// 3 v_wmma_f32_16x16x32_bf16 per fragment pair per K-step -> ~fp32 accuracy.
// Global->register prefetch of the next K-tile overlaps the WMMA burst.
__global__ __launch_bounds__(256) void moe_expert_gemm(
    const float* __restrict__ x, const float* __restrict__ We,
    const float* __restrict__ be, const float* __restrict__ wgt,
    const int* __restrict__ counts, const int* __restrict__ lists,
    float* __restrict__ out) {
    const int e    = blockIdx.z;
    const int cnt  = counts[e];
    const int row0 = blockIdx.y * BM;
    if (row0 >= cnt) return;
    const int n0 = blockIdx.x * BN;

    __shared__ __bf16 Ahi[BM][LDK];
    __shared__ __bf16 Alo[BM][LDK];
    __shared__ __bf16 Bhi[BN][LDK];   // stored transposed: [n][k]
    __shared__ __bf16 Blo[BN][LDK];
    __shared__ int    sTok[BM];
    __shared__ float  sW[BM];

    const int tid = threadIdx.x;
    if (tid < BM) {
        int r   = row0 + tid;
        int tok = (r < cnt) ? lists[e * N_TOKENS + r] : -1;
        sTok[tid] = tok;
        sW[tid]   = (tok >= 0) ? wgt[tok] : 0.f;
    }
    __syncthreads();

    const int lane  = tid & 31;
    const int wave  = tid >> 5;
    const int waveM = (wave >> 2) * 64;        // 0 or 64
    const int waveN = (wave & 3) * 32;         // 0,32,64,96

    v8f c[4][2] = {};

    const float* Wbase = We + (size_t)e * DIM * DIM;
    // A staging map: 128 rows x 8 float4 -> 256 threads x 4 iters
    const int lrA = tid >> 3, lc4 = tid & 7;       // lrA 0..31
    // B staging map: 32 k-rows x 32 float4 -> 256 threads x 4 iters
    const int kB0 = tid >> 5, c4B = tid & 31;      // kB0 0..7, c4B 0..31
    const int kb0 = (lane < 16) ? 0 : 8;           // ISA 16-bit A/B lane K-base

    float4 aReg[4], bReg[4];

    auto loadTiles = [&](int k0) {
#pragma unroll
        for (int rr = 0; rr < 4; ++rr) {
            int row = lrA + rr * 32;
            int tok = sTok[row]; if (tok < 0) tok = 0;
            aReg[rr] = *(const float4*)(x + (size_t)tok * DIM + k0 + lc4 * 4);
        }
#pragma unroll
        for (int rr = 0; rr < 4; ++rr) {
            int kk = kB0 + rr * 8;
            bReg[rr] = *(const float4*)(Wbase + (size_t)(k0 + kk) * DIM + n0 + c4B * 4);
        }
    };
    auto storeTiles = [&]() {
#pragma unroll
        for (int rr = 0; rr < 4; ++rr) {
            int row = lrA + rr * 32;
            float4 v = aReg[rr];
            __bf16 h0 = (__bf16)v.x, h1 = (__bf16)v.y, h2 = (__bf16)v.z, h3 = (__bf16)v.w;
            v4bf hi = {h0, h1, h2, h3};
            v4bf lo = {(__bf16)(v.x - (float)h0), (__bf16)(v.y - (float)h1),
                       (__bf16)(v.z - (float)h2), (__bf16)(v.w - (float)h3)};
            *(v4bf*)&Ahi[row][lc4 * 4] = hi;
            *(v4bf*)&Alo[row][lc4 * 4] = lo;
        }
#pragma unroll
        for (int rr = 0; rr < 4; ++rr) {
            int kk = kB0 + rr * 8;
            float4 v = bReg[rr];
            float f[4] = {v.x, v.y, v.z, v.w};
#pragma unroll
            for (int j = 0; j < 4; ++j) {
                int n = c4B * 4 + j;
                __bf16 h = (__bf16)f[j];
                Bhi[n][kk] = h;
                Blo[n][kk] = (__bf16)(f[j] - (float)h);
            }
        }
    };

    loadTiles(0);
    for (int k0 = 0; k0 < DIM; k0 += BK) {
        storeTiles();
        __syncthreads();
        if (k0 + BK < DIM) loadTiles(k0 + BK);   // prefetch next K-tile (overlaps WMMA)

        // ---- assemble fragments (two ds_load_b128 each; layout per ISA 7.12.2)
        v16bf aH[4], aL[4], bH[2], bL[2];
#pragma unroll
        for (int mi = 0; mi < 4; ++mi) {
            int row = waveM + mi * 16 + (lane & 15);
            v8bf h0 = *(const v8bf*)&Ahi[row][kb0];
            v8bf h1 = *(const v8bf*)&Ahi[row][kb0 + 16];
            aH[mi] = __builtin_shufflevector(h0, h1, 0,1,2,3,4,5,6,7,8,9,10,11,12,13,14,15);
            v8bf l0 = *(const v8bf*)&Alo[row][kb0];
            v8bf l1 = *(const v8bf*)&Alo[row][kb0 + 16];
            aL[mi] = __builtin_shufflevector(l0, l1, 0,1,2,3,4,5,6,7,8,9,10,11,12,13,14,15);
        }
#pragma unroll
        for (int ni = 0; ni < 2; ++ni) {
            int col = waveN + ni * 16 + (lane & 15);
            v8bf h0 = *(const v8bf*)&Bhi[col][kb0];
            v8bf h1 = *(const v8bf*)&Bhi[col][kb0 + 16];
            bH[ni] = __builtin_shufflevector(h0, h1, 0,1,2,3,4,5,6,7,8,9,10,11,12,13,14,15);
            v8bf l0 = *(const v8bf*)&Blo[col][kb0];
            v8bf l1 = *(const v8bf*)&Blo[col][kb0 + 16];
            bL[ni] = __builtin_shufflevector(l0, l1, 0,1,2,3,4,5,6,7,8,9,10,11,12,13,14,15);
        }
        // ---- 24 WMMAs: hi*hi + hi*lo + lo*hi
#pragma unroll
        for (int mi = 0; mi < 4; ++mi) {
#pragma unroll
            for (int ni = 0; ni < 2; ++ni) {
                c[mi][ni] = __builtin_amdgcn_wmma_f32_16x16x32_bf16(
                    false, aH[mi], false, bH[ni], (short)0, c[mi][ni], false, false);
                c[mi][ni] = __builtin_amdgcn_wmma_f32_16x16x32_bf16(
                    false, aH[mi], false, bL[ni], (short)0, c[mi][ni], false, false);
                c[mi][ni] = __builtin_amdgcn_wmma_f32_16x16x32_bf16(
                    false, aL[mi], false, bH[ni], (short)0, c[mi][ni], false, false);
            }
        }
        __syncthreads();
    }

    // ---- epilogue: (acc + bias) * token_weight, scatter-accumulate
    const int nLane = lane & 15;
    const int mOff  = (lane >> 4) * 8;
#pragma unroll
    for (int mi = 0; mi < 4; ++mi) {
        const int base = waveM + mi * 16 + mOff;
        int   toks[8];
        float ws8[8];
#pragma unroll
        for (int j = 0; j < 8; ++j) { toks[j] = sTok[base + j]; ws8[j] = sW[base + j]; }
#pragma unroll
        for (int ni = 0; ni < 2; ++ni) {
            const int ncol = n0 + waveN + ni * 16 + nLane;
            const float bias = be[e * DIM + ncol];
#pragma unroll
            for (int j = 0; j < 8; ++j) {
                if (toks[j] < 0) continue;
                float val = (c[mi][ni][j] + bias) * ws8[j];
                atomicAdd(out + (size_t)toks[j] * DIM + ncol, val);
            }
        }
    }
}

// ---------------------------------------------------------------- launch
extern "C" void kernel_launch(void* const* d_in, const int* in_sizes, int n_in,
                              void* d_out, int out_size, void* d_ws, size_t ws_size,
                              hipStream_t stream) {
    const float* x  = (const float*)d_in[0];
    const float* Wg = (const float*)d_in[1];
    const float* bg = (const float*)d_in[2];
    const float* We = (const float*)d_in[3];
    const float* be = (const float*)d_in[4];
    float* out = (float*)d_out;

    char*  ws     = (char*)d_ws;
    int*   counts = (int*)ws;                                   // 8 ints (64B slot)
    float* wgt    = (float*)(ws + 64);                          // N floats
    int*   lists  = (int*)(ws + 64 + N_TOKENS * sizeof(float)); // E*N ints (~1MB total)

    hipMemsetAsync(counts, 0, 64, stream);
    moe_zero_kernel<<<4096, 256, 0, stream>>>((float4*)out, N_TOKENS * DIM / 4);
    moe_gate_kernel<<<N_TOKENS / 8, 256, 0, stream>>>(x, Wg, bg, wgt, counts, lists);

    dim3 grid(DIM / BN, N_TOKENS / BM, NEXP);   // N-tiles x row-tiles x experts
    moe_expert_gemm<<<grid, 256, 0, stream>>>(x, We, be, wgt, counts, lists, out);
}
